// Attention_50551765074448
// MI455X (gfx1250) — compile-verified, matrix operand order checked by reading, not run on hardware
//
#include <hip/hip_runtime.h>
#include <stdint.h>

#define S_LEN 4096
#define HDIM  1024
#define NHEAD 16
#define DHEAD 64
#define KSEL  32

typedef __attribute__((ext_vector_type(16))) __bf16 v16bf;
typedef __attribute__((ext_vector_type(8)))  __bf16 v8bf;
typedef __attribute__((ext_vector_type(8)))  float  v8f;

#if defined(__has_builtin)
#  if __has_builtin(__builtin_amdgcn_global_load_async_to_lds_b128)
#    define HAVE_ASYNC_LDS 1
#  endif
#endif
#ifndef HAVE_ASYNC_LDS
#  define HAVE_ASYNC_LDS 0
#endif

#if defined(__has_builtin)
#  if __has_builtin(__builtin_amdgcn_s_wait_asynccnt)
#    define WAIT_ASYNC(n) __builtin_amdgcn_s_wait_asynccnt(n)
#  endif
#endif
#ifndef WAIT_ASYNC
#  define WAIT_ASYNC(n) asm volatile("s_wait_asynccnt %0" :: "i"(n) : "memory")
#endif

// Builtin expects int4 (vector_size(16)) pointers: AS1 global src, AS3 LDS dst.
typedef int v4i_gcc __attribute__((vector_size(16)));
typedef __attribute__((address_space(1))) v4i_gcc as1_v4i;
typedef __attribute__((address_space(3))) v4i_gcc as3_v4i;

static __device__ __forceinline__ uint16_t f32_to_bf16_rne(float f) {
    uint32_t u = __float_as_uint(f);
    uint32_t r = u + 0x7FFFu + ((u >> 16) & 1u);
    return (uint16_t)(r >> 16);
}

// ---------------------------------------------------------------- convert
__global__ void cvt_f32_bf16(const float* __restrict__ in,
                             uint16_t* __restrict__ out, int n) {
    int i = blockIdx.x * blockDim.x + threadIdx.x;
    int stride = gridDim.x * blockDim.x;
    for (; i < n; i += stride) out[i] = f32_to_bf16_rne(in[i]);
}

// ------------------------------------------------------------- WMMA GEMM
// Fragment loader (from LDS). Per ISA 7.12.2 for 16-bit A/B 16x32 frags:
// lanes 0-15 hold K=0..7 / 16..23, lanes 16-31 hold K=8..15 / 24..31.
__device__ __forceinline__ v16bf load_frag(const uint16_t* p) {
    v8bf lo = *reinterpret_cast<const v8bf*>(p);        // K = kb .. kb+7
    v8bf hi = *reinterpret_cast<const v8bf*>(p + 16);   // K = kb+16 .. kb+23
    v16bf o;
#pragma unroll
    for (int i = 0; i < 8; ++i) { o[i] = lo[i]; o[i + 8] = hi[i]; }
    return o;
}

#define LROW 40   // LDS slab row pitch in uint16 (80 B: 64 B data + pad)

// C[M,N] (f32) = A[M,Kd](bf16) * Bt[N,Kd]^T(bf16) + bias.
// 128-thread block = 4 waves (2x2), block tile 128x128, wave tile 64x64
// (4x4 WMMA tiles). A/B K-slabs (128x32 bf16) double-buffered in LDS,
// filled with ASYNCcnt-tracked global_load_async_to_lds_b128.
__global__ __launch_bounds__(128)
void gemm_bf16_wmma(const uint16_t* __restrict__ A,
                    const uint16_t* __restrict__ Bt,
                    float* __restrict__ C,
                    const float* __restrict__ bias,
                    int M, int N, int Kd) {
    __shared__ uint16_t lsA[2][128 * LROW];
    __shared__ uint16_t lsB[2][128 * LROW];

    const int tid  = threadIdx.x;
    const int lane = tid & 31;
    const int w    = tid >> 5;                 // 0..3
    const int blockRow = blockIdx.y * 128;
    const int blockCol = blockIdx.x * 128;
    const int row0 = (w & 1) * 64;             // wave tile origin in block
    const int col0 = (w >> 1) * 64;

    // Slab fill: 512 x 16B chunks per slab, 4 per thread per slab.
    auto fill = [&](int buf, int kk) {
#pragma unroll
        for (int t = 0; t < 4; ++t) {
            const int c    = tid + (t << 7);       // 0..511
            const int row  = c >> 2;               // slab row
            const int part = c & 3;                // 16B chunk within 64B
            const uint16_t* ga = A  + (size_t)(blockRow + row) * Kd + kk + part * 8;
            const uint16_t* gb = Bt + (size_t)(blockCol + row) * Kd + kk + part * 8;
            uint16_t* la = &lsA[buf][row * LROW + part * 8];
            uint16_t* lb = &lsB[buf][row * LROW + part * 8];
#if HAVE_ASYNC_LDS
            __builtin_amdgcn_global_load_async_to_lds_b128(
                (as1_v4i*)(void*)ga, (as3_v4i*)(void*)la, 0, 0);
            __builtin_amdgcn_global_load_async_to_lds_b128(
                (as1_v4i*)(void*)gb, (as3_v4i*)(void*)lb, 0, 0);
#else
            *(uint4*)la = *(const uint4*)ga;
            *(uint4*)lb = *(const uint4*)gb;
#endif
        }
    };

    const int fr = lane & 15;           // M (A) / N (B) within frag
    const int fk = (lane >> 4) << 3;    // 0 or 8 (K half)

    v8f zero = {};
    v8f acc[4][4];
#pragma unroll
    for (int mi = 0; mi < 4; ++mi)
#pragma unroll
        for (int ni = 0; ni < 4; ++ni) acc[mi][ni] = zero;

    fill(0, 0);                          // prologue: first slab in flight
    int ib = 0;
#pragma unroll 1
    for (int kk = 0; kk < Kd; kk += 32, ib ^= 1) {
        const bool more = (kk + 32) < Kd;
        if (more) fill(ib ^ 1, kk + 32); // next slab overlaps compute
#if HAVE_ASYNC_LDS
        if (more) WAIT_ASYNC(8);         // older batch (8/thread) complete
        else      WAIT_ASYNC(0);
#endif
        __syncthreads();

        v16bf af[4], bfr[4];
#pragma unroll
        for (int t = 0; t < 4; ++t) {
            af[t]  = load_frag(&lsA[ib][(row0 + t * 16 + fr) * LROW + fk]);
            bfr[t] = load_frag(&lsB[ib][(col0 + t * 16 + fr) * LROW + fk]);
        }
#pragma unroll
        for (int mi = 0; mi < 4; ++mi)
#pragma unroll
            for (int ni = 0; ni < 4; ++ni)
                acc[mi][ni] = __builtin_amdgcn_wmma_f32_16x16x32_bf16(
                    false, af[mi], false, bfr[ni], (short)0, acc[mi][ni],
                    false, false);
        __syncthreads();                 // safe to refill this buffer
    }

    // C/D layout: lane&15 = N, VGPR r = M (+8 for lanes 16-31)
    const int nloc = lane & 15;
    const int mhi  = (lane >> 4) << 3;
#pragma unroll
    for (int ni = 0; ni < 4; ++ni) {
        const int cb = blockCol + col0 + ni * 16 + nloc;
        const float bv = bias ? bias[cb] : 0.0f;
#pragma unroll
        for (int mi = 0; mi < 4; ++mi) {
#pragma unroll
            for (int r = 0; r < 8; ++r) {
                C[(size_t)(blockRow + row0 + mi * 16 + mhi + r) * N + cb] =
                    acc[mi][ni][r] + bv;
            }
        }
    }
}

// ----------------------------------------------------- gathered attention
// One wave32 per (s, h). Lane k handles selected key k (K=32 == wave width).
__global__ __launch_bounds__(256)
void attn_gather(const float* __restrict__ q,
                 const float* __restrict__ kmat,
                 const float* __restrict__ vmat,
                 const int* __restrict__ idx,
                 const unsigned char* __restrict__ valid,
                 const float* __restrict__ geo_bias,
                 uint16_t* __restrict__ outbf) {
    __shared__ float qsh[8][DHEAD];
    const int lane = threadIdx.x & 31;
    const int w    = threadIdx.x >> 5;
    const int g    = blockIdx.x * 8 + w;     // (s * NHEAD + h)
    const int s    = g >> 4;
    const int h    = g & 15;

    const float* qrow = q + (size_t)s * HDIM + h * DHEAD;
    qsh[w][lane]      = qrow[lane];
    qsh[w][lane + 32] = qrow[lane + 32];
    __syncthreads();

    const int  j  = idx[(size_t)s * KSEL + lane];
    const bool ok = (j <= s) && (valid[(size_t)s * KSEL + lane] != 0);

    // logit_k = (q . k_sel[j]) / sqrt(64) + geo_bias[h,s,k]
    const float* krow = kmat + (size_t)j * HDIM + h * DHEAD;
    float acc = 0.0f;
#pragma unroll
    for (int d = 0; d < DHEAD; d += 4) {
        const float4 kv = *reinterpret_cast<const float4*>(krow + d);
        acc += kv.x * qsh[w][d]     + kv.y * qsh[w][d + 1]
             + kv.z * qsh[w][d + 2] + kv.w * qsh[w][d + 3];
    }
    float logit = acc * 0.125f +
                  geo_bias[((size_t)h * S_LEN + s) * KSEL + lane];
    logit = ok ? logit : -3.0e38f;

    // wave32 softmax over the 32 lanes
    float m = logit;
#pragma unroll
    for (int off = 16; off > 0; off >>= 1) m = fmaxf(m, __shfl_xor(m, off, 32));
    float e = ok ? __expf(logit - m) : 0.0f;
    float sum = e;
#pragma unroll
    for (int off = 16; off > 0; off >>= 1) sum += __shfl_xor(sum, off, 32);
    const float wgt = e / sum;

    // out[d] = sum_k w_k * v[j_k][d]; lanes cooperatively own d, d+32
    float o0 = 0.0f, o1 = 0.0f;
#pragma unroll
    for (int kq = 0; kq < KSEL; ++kq) {
        const float wk = __shfl(wgt, kq, 32);
        const int   jk = __shfl(j,   kq, 32);
        const float* vrow = vmat + (size_t)jk * HDIM + h * DHEAD;
        o0 += wk * vrow[lane];
        o1 += wk * vrow[lane + 32];
    }
    uint16_t* orow = outbf + (size_t)s * HDIM + h * DHEAD;
    orow[lane]      = f32_to_bf16_rne(o0);
    orow[lane + 32] = f32_to_bf16_rne(o1);
}

// ---------------------------------------------------------------- launch
extern "C" void kernel_launch(void* const* d_in, const int* in_sizes, int n_in,
                              void* d_out, int out_size, void* d_ws, size_t ws_size,
                              hipStream_t stream) {
    const float*         x     = (const float*)d_in[0];
    const int*           idx   = (const int*)d_in[1];
    const unsigned char* valid = (const unsigned char*)d_in[2];
    const float*         geo   = (const float*)d_in[3];
    const float*         Wq    = (const float*)d_in[4];
    const float*         Wk    = (const float*)d_in[5];
    const float*         Wv    = (const float*)d_in[6];
    const float*         Wo    = (const float*)d_in[7];
    const float*         bo    = (const float*)d_in[8];
    float*               out   = (float*)d_out;

    char* ws = (char*)d_ws;
    uint16_t* xbf  = (uint16_t*)(ws);                       //  8 MiB
    uint16_t* wqbf = (uint16_t*)(ws + ((size_t)8  << 20));  //  2 MiB
    uint16_t* wkbf = (uint16_t*)(ws + ((size_t)10 << 20));
    uint16_t* wvbf = (uint16_t*)(ws + ((size_t)12 << 20));
    uint16_t* wobf = (uint16_t*)(ws + ((size_t)14 << 20));
    float*    qf   = (float*)   (ws + ((size_t)16 << 20));  // 16 MiB
    float*    kf   = (float*)   (ws + ((size_t)32 << 20));
    float*    vf   = (float*)   (ws + ((size_t)48 << 20));
    uint16_t* abf  = (uint16_t*)(ws + ((size_t)64 << 20));  //  8 MiB

    const int NX = S_LEN * HDIM;   // 4,194,304
    const int NW = HDIM * HDIM;    // 1,048,576

    cvt_f32_bf16<<<1024, 256, 0, stream>>>(x,  xbf,  NX);
    cvt_f32_bf16<<<512,  256, 0, stream>>>(Wq, wqbf, NW);
    cvt_f32_bf16<<<512,  256, 0, stream>>>(Wk, wkbf, NW);
    cvt_f32_bf16<<<512,  256, 0, stream>>>(Wv, wvbf, NW);
    cvt_f32_bf16<<<512,  256, 0, stream>>>(Wo, wobf, NW);

    dim3 gg(HDIM / 128, S_LEN / 128);   // 8 x 32 blocks of 128 threads
    gemm_bf16_wmma<<<gg, 128, 0, stream>>>(xbf, wqbf, qf, nullptr, S_LEN, HDIM, HDIM);
    gemm_bf16_wmma<<<gg, 128, 0, stream>>>(xbf, wkbf, kf, nullptr, S_LEN, HDIM, HDIM);
    gemm_bf16_wmma<<<gg, 128, 0, stream>>>(xbf, wvbf, vf, nullptr, S_LEN, HDIM, HDIM);

    attn_gather<<<(NHEAD * S_LEN) / 8, 256, 0, stream>>>(qf, kf, vf, idx, valid,
                                                         geo, abf);

    gemm_bf16_wmma<<<gg, 128, 0, stream>>>(abf, wobf, out, bo, S_LEN, HDIM, HDIM);
}
